// ContainedLSTM_6408091206252
// MI455X (gfx1250) — compile-verified, matrix operand order checked by reading, not run on hardware
//
#include <hip/hip_runtime.h>
#include <hip/hip_bf16.h>

typedef __attribute__((ext_vector_type(16))) __bf16 v16bf;
typedef __attribute__((ext_vector_type(8)))  float  v8f;

#define HSZ   256
#define ISZ   256
#define BSZ   128
#define SSZ   1024
#define G4H   1024          // 4*H
#define LN_EPS 1e-5f

// ---- WMMA 16x16x32 bf16 operand layouts (CDNA5 ISA 7.12.2) -----------------
// A (16x32, MxK): lane L (L<16) holds row M=L, K in {0..7,16..23};
//                 lane L+16 holds row M=L, K in {8..15,24..31}.
// element e of v16bf: k = (e>>3)*16 + (L>>4)*8 + (e&7)
// B (32x16, KxN) mirrors A with M<->N.
// C/D f32 16x16: lane L -> col n=L&15, VGPR r -> row m = r + 8*(L>>4).
__device__ __forceinline__ void a_layout(int m, int k, int& lane, int& elem) {
    int kk = k & 31;
    lane = m + ((((kk & 15) >= 8)) ? 16 : 0);
    elem = ((kk >= 16) ? 8 : 0) + (kk & 7);
}

// Pack W (4H x 256 f32, row-major) into bf16 B-operand order:
// Wp[((nt*8 + kb)*32 + lane)*16 + e]  with  n = nt*16 + (lane&15),
// k = kb*32 + (e>>3)*16 + (lane>>4)*8 + (e&7)
__global__ void pack_weights(const float* __restrict__ W, __bf16* __restrict__ Wp) {
    int idx = blockIdx.x * blockDim.x + threadIdx.x;     // 64*8*32*16 = 262144
    if (idx >= 64 * 8 * 32 * 16) return;
    int e  = idx & 15;
    int L  = (idx >> 4) & 31;
    int kb = (idx >> 9) & 7;
    int nt = idx >> 12;
    int n  = nt * 16 + (L & 15);
    int k  = kb * 32 + ((e >> 3) * 16) + ((L >> 4) * 8) + (e & 7);
    Wp[idx] = (__bf16)W[n * 256 + k];
}

// Persistent LSTM: one workgroup per 16 batch rows, loops over all S steps.
// 512 threads = 16 waves; wave w owns N-tiles [4w, 4w+4) of the 4H=1024 gates.
__global__ __launch_bounds__(512) void lstm_persistent(
    const float* __restrict__ x, const int* __restrict__ mask,
    const __bf16* WihP, const __bf16* WhhP,
    const float* __restrict__ b_ih, const float* __restrict__ b_hh,
    const float* __restrict__ gamma, const float* __restrict__ beta,
    float* __restrict__ out) {

    extern __shared__ __align__(64) char smem[];
    float*  gates = (float*)smem;                       // 16*1024 f32 = 64 KB
    float*  hS    = gates + 16 * G4H;                   // 16 KB
    float*  cS    = hS + 16 * HSZ;                      // 16 KB
    float*  oS    = cS + 16 * HSZ;                      // 16 KB
    __bf16* hnA   = (__bf16*)(oS + 16 * HSZ);           // 8 KB
    __bf16* xA    = hnA + 8 * 32 * 16;                  // 8 KB
    float*  gamS  = (float*)(xA + 8 * 32 * 16);         // 1 KB
    float*  betS  = gamS + HSZ;                         // 1 KB  (total 130 KB)

    const int tid  = threadIdx.x;
    const int wave = tid >> 5;
    const int lane = tid & 31;
    const int b0   = blockIdx.x * 16;

    for (int i = tid; i < 16 * HSZ; i += 512) { hS[i] = 0.f; cS[i] = 0.f; oS[i] = 0.f; }
    if (tid < HSZ) { gamS[tid] = gamma[tid]; betS[tid] = beta[tid]; }

    int   ntile[4];
    float bias[4];
    for (int u = 0; u < 4; ++u) {
        ntile[u] = wave * 4 + u;
        int n = ntile[u] * 16 + (lane & 15);
        bias[u] = b_ih[n] + b_hh[n];
    }
    __syncthreads();

#pragma clang loop unroll(disable)
    for (int t = 0; t < SSZ; ++t) {
        // Forbid hoisting the t-invariant weight loads out of the step loop.
        asm volatile("" ::: "memory");

        // ---- Phase 1a: load x_t tile -> bf16 A-layout in LDS ----
        for (int idx = tid; idx < 16 * ISZ; idx += 512) {
            int m = idx >> 8, k = idx & 255;
            const float* xp = x + (((size_t)(b0 + m)) * SSZ + t) * ISZ + k;
            float xv = *xp;
            if (t + 1 < SSZ) __builtin_prefetch(xp + ISZ, 0, 1);  // next step's tile
            int dl, de; a_layout(m, k, dl, de);
            xA[((k >> 5) * 32 + dl) * 16 + de] = (__bf16)xv;
        }
        // ---- Phase 1b: layernorm(h) row `wave` -> bf16 A-layout ----
        {
            int m = wave;
            float v[8], s = 0.f, sq = 0.f;
            for (int j = 0; j < 8; ++j) {
                float hv = hS[m * HSZ + lane + 32 * j];
                v[j] = hv; s += hv; sq += hv * hv;
            }
            for (int off = 16; off >= 1; off >>= 1) {
                s  += __shfl_xor(s,  off, 32);
                sq += __shfl_xor(sq, off, 32);
            }
            float mu   = s * (1.0f / HSZ);
            float var  = sq * (1.0f / HSZ) - mu * mu;
            float rstd = rsqrtf(var + LN_EPS);
            for (int j = 0; j < 8; ++j) {
                int k = lane + 32 * j;
                float hn = (v[j] - mu) * rstd * gamS[k] + betS[k];
                int dl, de; a_layout(m, k, dl, de);
                hnA[((k >> 5) * 32 + dl) * 16 + de] = (__bf16)hn;
            }
        }

        // ---- Phase 2 prologue: kb=0 weight loads issued BEFORE the barrier,
        // so they are in flight across phase-1 LDS work + s_barrier. ----
        v8f acc[4];
#pragma unroll
        for (int u = 0; u < 4; ++u)
#pragma unroll
            for (int r = 0; r < 8; ++r) acc[u][r] = bias[u];

        v16bf bX[4], bH[4];
#pragma unroll
        for (int u = 0; u < 4; ++u) {
            bX[u] = *(const v16bf*)&WihP[((ntile[u] * 8) * 32 + lane) * 16];
            bH[u] = *(const v16bf*)&WhhP[((ntile[u] * 8) * 32 + lane) * 16];
        }
        __syncthreads();   // xA/hnA now visible; B(kb=0) still in flight

        v16bf aX = *(const v16bf*)&xA [(lane) * 16];
        v16bf aH = *(const v16bf*)&hnA[(lane) * 16];

        // ---- Phase 2: software-pipelined WMMA over K ----
#pragma unroll
        for (int kb = 0; kb < 8; ++kb) {
            v16bf naX, naH, nbX[4], nbH[4];
            if (kb < 7) {
#pragma unroll
                for (int u = 0; u < 4; ++u) {
                    nbX[u] = *(const v16bf*)&WihP[((ntile[u] * 8 + kb + 1) * 32 + lane) * 16];
                    nbH[u] = *(const v16bf*)&WhhP[((ntile[u] * 8 + kb + 1) * 32 + lane) * 16];
                }
                naX = *(const v16bf*)&xA [((kb + 1) * 32 + lane) * 16];
                naH = *(const v16bf*)&hnA[((kb + 1) * 32 + lane) * 16];
            }
#pragma unroll
            for (int u = 0; u < 4; ++u)
                acc[u] = __builtin_amdgcn_wmma_f32_16x16x32_bf16(
                    false, aX, false, bX[u], (short)0, acc[u], false, false);
#pragma unroll
            for (int u = 0; u < 4; ++u)
                acc[u] = __builtin_amdgcn_wmma_f32_16x16x32_bf16(
                    false, aH, false, bH[u], (short)0, acc[u], false, false);
            if (kb < 7) {
                aX = naX; aH = naH;
#pragma unroll
                for (int u = 0; u < 4; ++u) { bX[u] = nbX[u]; bH[u] = nbH[u]; }
            }
        }
        {
            int mh = (lane >> 4) * 8;
            for (int u = 0; u < 4; ++u) {
                int n = ntile[u] * 16 + (lane & 15);
                for (int r = 0; r < 8; ++r)
                    gates[(mh + r) * G4H + n] = acc[u][r];
            }
        }
        __syncthreads();

        // ---- Phase 3: LSTM cell + masked output select ----
        for (int e = 0; e < 8; ++e) {
            int id = tid + e * 512;          // 0..4095
            int m  = id >> 8;
            int j  = id & 255;
            float ig = gates[m * G4H + j];
            float fg = gates[m * G4H + HSZ + j];
            float gg = gates[m * G4H + 2 * HSZ + j];
            float og = gates[m * G4H + 3 * HSZ + j];
            float si = 1.0f / (1.0f + __expf(-ig));
            float sf = 1.0f / (1.0f + __expf(-fg));
            float so = 1.0f / (1.0f + __expf(-og));
            float c  = sf * cS[id] + si * tanhf(gg);
            float h  = so * tanhf(c);
            cS[id] = c;
            hS[id] = h;
            if (!mask[(size_t)(b0 + m) * SSZ + t]) oS[id] = h;
        }
        __syncthreads();
    }

    for (int e = 0; e < 8; ++e) {
        int id = tid + e * 512;
        out[(size_t)(b0 + (id >> 8)) * HSZ + (id & 255)] = oS[id];
    }
}

extern "C" void kernel_launch(void* const* d_in, const int* in_sizes, int n_in,
                              void* d_out, int out_size, void* d_ws, size_t ws_size,
                              hipStream_t stream) {
    const float* x     = (const float*)d_in[0];
    const int*   mask  = (const int*)  d_in[1];
    const float* W_ih  = (const float*)d_in[2];
    const float* W_hh  = (const float*)d_in[3];
    const float* b_ih  = (const float*)d_in[4];
    const float* b_hh  = (const float*)d_in[5];
    const float* gamma = (const float*)d_in[6];
    const float* beta  = (const float*)d_in[7];
    float* out = (float*)d_out;

    __bf16* WihP = (__bf16*)d_ws;                       // 512 KB
    __bf16* WhhP = WihP + 64 * 8 * 32 * 16;             // +512 KB

    pack_weights<<<1024, 256, 0, stream>>>(W_ih, WihP);
    pack_weights<<<1024, 256, 0, stream>>>(W_hh, WhhP);

    const size_t smemBytes =
        (16 * G4H + 3 * 16 * HSZ + 2 * HSZ) * sizeof(float) +
        2 * (8 * 32 * 16) * sizeof(__bf16);             // 133,120 B < 320 KB/WGP
    (void)hipFuncSetAttribute((const void*)lstm_persistent,
                              hipFuncAttributeMaxDynamicSharedMemorySize,
                              (int)smemBytes);
    lstm_persistent<<<BSZ / 16, 512, smemBytes, stream>>>(
        x, mask, WihP, WhhP, b_ih, b_hh, gamma, beta, out);
}
